// MLA_18751827214733
// MI455X (gfx1250) — compile-verified, matrix operand order checked by reading, not run on hardware
//
#include <hip/hip_runtime.h>
#include <hip/hip_bf16.h>
#include <math.h>

#define S_LEN  2048
#define BSZ    2
#define NH     8
#define DHEAD  64
#define RDIM   8
#define HPAD   96               // 72 (DH+R) padded to 3x32 WMMA K-steps
#define BS_ROWS (BSZ * S_LEN)   // 4096

typedef __attribute__((ext_vector_type(16))) __bf16          v16bf;
typedef __attribute__((ext_vector_type(8)))  float           v8f;
typedef __attribute__((ext_vector_type(8)))  unsigned short  u16x8;

union FragAB {
    v16bf v;
    u16x8 h[2];
    unsigned short s[16];
};

__device__ __forceinline__ unsigned short f2bf(float f) {
    union { float f; unsigned int u; } cv; cv.f = f;
    unsigned int u = cv.u;
    return (unsigned short)((u + 0x7FFFu + ((u >> 16) & 1u)) >> 16);   // RNE
}

__device__ __forceinline__ v8f zero8() {
    v8f z;
    #pragma unroll
    for (int i = 0; i < 8; i++) z[i] = 0.0f;
    return z;
}

// ---- CDNA5 async copy: global -> LDS, 16B per lane, ASYNCcnt tracked ------
__device__ __forceinline__ void async_b128(const void* gptr, const void* lptr) {
    unsigned lds = (unsigned)(uintptr_t)lptr;       // low 32 bits = LDS offset
    unsigned long long ga = (unsigned long long)(uintptr_t)gptr;
    asm volatile("global_load_async_to_lds_b128 %0, %1, off"
                 :: "v"(lds), "v"(ga) : "memory");
}
#define WAIT_ASYNC(n) asm volatile("s_wait_asynccnt " #n ::: "memory")
#define WAIT_DS0()    asm volatile("s_wait_dscnt 0x0" ::: "memory")

// Load a 16x32 bf16 WMMA fragment (ISA 7.12.2 per-lane layout) from an LDS
// tile stored [rows][stride] (ushort): two 16B ds_load_b128 per lane.
__device__ __forceinline__ v16bf load_frag(const unsigned short* base, int row,
                                           int stride, int coloff, int lane) {
    int k0 = ((lane >> 4) & 1) * 8;
    FragAB f;
    f.h[0] = *(const u16x8*)(base + row * stride + coloff + k0);
    f.h[1] = *(const u16x8*)(base + row * stride + coloff + 16 + k0);
    return f.v;
}

__device__ __forceinline__ v8f wmma_bf16(v16bf a, v16bf b, v8f c) {
    return __builtin_amdgcn_wmma_f32_16x16x32_bf16(false, a, false, b,
                                                   (short)0, c, false, false);
}

// ---------------------------------------------------------------------------
// Weight converter: W (fp32) -> Wt (bf16) laid out [Npad][Kpad], zero padded.
// srcKN=1: W stored [K][N] (transpose). srcKN=0: W stored [N][K] (W_o case).
// ---------------------------------------------------------------------------
__global__ void convW_kernel(const float* __restrict__ W,
                             unsigned short* __restrict__ Wt,
                             int K, int N, int Kpad, int total, int srcKN)
{
    int idx = blockIdx.x * 256 + threadIdx.x;
    if (idx >= total) return;
    int n = idx / Kpad, k = idx % Kpad;
    float v = 0.0f;
    if (n < N && k < K)
        v = srcKN ? W[(size_t)k * N + n] : W[(size_t)n * K + k];
    Wt[idx] = f2bf(v);
}

// Activation converter: X (fp32 [M][K]) -> Xb (bf16 [M][Kpad], zero padded K)
__global__ void convA_kernel(const float* __restrict__ X,
                             unsigned short* __restrict__ Xb,
                             int K, int Kpad, int total)
{
    int idx = blockIdx.x * 256 + threadIdx.x;
    if (idx >= total) return;
    int m = idx / Kpad, k = idx % Kpad;
    Xb[idx] = f2bf((k < K) ? X[(size_t)m * K + k] : 0.0f);
}

// ---------------------------------------------------------------------------
// GEMM: C[M,Nout](fp32) = A[M,K](bf16) @ Bt[Npad,K](bf16, pre-transposed).
// 256 threads = 8 waves (2x4); wave tile 32x32 (4 wmma / K-step);
// workgroup tile 64x128. Double-buffered async tile loads. K % 32 == 0,
// M % 64 == 0, Npad % 128 == 0 (only the C store is guarded).
// ---------------------------------------------------------------------------
__global__ void gemm_async_kernel(const unsigned short* __restrict__ A,
                                  const unsigned short* __restrict__ Bt,
                                  float* __restrict__ C,
                                  int K, int Nout, int ldc)
{
    __shared__ __align__(16) unsigned short sA[2][64 * 32];
    __shared__ __align__(16) unsigned short sB[2][128 * 32];

    const int t    = threadIdx.x;
    const int lane = t & 31;
    const int w    = t >> 5;
    const int wr   = w >> 2;            // 0..1
    const int wc   = w & 3;             // 0..3
    const int m0   = blockIdx.y * 64;
    const int n0   = blockIdx.x * 128;
    const int aRow = lane & 15;
    const int hi   = ((lane >> 4) & 1) * 8;

    // per-thread async chunk assignment (16B chunks)
    const int ar = t >> 2, ac = t & 3;          // A: 64 rows x 4 chunks

    auto issue_tile = [&](int k0, int buf) {
        async_b128(A + (size_t)(m0 + ar) * K + k0 + ac * 8,
                   &sA[buf][ar * 32 + ac * 8]);
        #pragma unroll
        for (int i = 0; i < 2; i++) {           // B: 128 rows x 4 chunks
            int id = i * 256 + t;
            int r = id >> 2, ch = id & 3;
            async_b128(Bt + (size_t)(n0 + r) * K + k0 + ch * 8,
                       &sB[buf][r * 32 + ch * 8]);
        }
    };

    v8f acc[2][2];
    #pragma unroll
    for (int i = 0; i < 2; i++)
        #pragma unroll
        for (int j = 0; j < 2; j++) acc[i][j] = zero8();

    const int nk = K >> 5;
    issue_tile(0, 0);
    for (int it = 0; it < nk; it++) {
        if (it + 1 < nk) {
            issue_tile((it + 1) << 5, (it + 1) & 1);
            WAIT_ASYNC(3);                      // older tile's 3 loads done
        } else {
            WAIT_ASYNC(0);
        }
        __syncthreads();

        const unsigned short* a  = sA[it & 1];
        const unsigned short* bt = sB[it & 1];
        v16bf fa0 = load_frag(a,  wr * 32 + aRow,      32, 0, lane);
        v16bf fa1 = load_frag(a,  wr * 32 + 16 + aRow, 32, 0, lane);
        v16bf fb0 = load_frag(bt, wc * 32 + aRow,      32, 0, lane);
        v16bf fb1 = load_frag(bt, wc * 32 + 16 + aRow, 32, 0, lane);
        acc[0][0] = wmma_bf16(fa0, fb0, acc[0][0]);
        acc[0][1] = wmma_bf16(fa0, fb1, acc[0][1]);
        acc[1][0] = wmma_bf16(fa1, fb0, acc[1][0]);
        acc[1][1] = wmma_bf16(fa1, fb1, acc[1][1]);
        __syncthreads();
    }

    #pragma unroll
    for (int j = 0; j < 2; j++) {
        int gn = n0 + wc * 32 + j * 16 + (lane & 15);
        if (gn >= Nout) continue;
        #pragma unroll
        for (int i = 0; i < 2; i++)
            #pragma unroll
            for (int r = 0; r < 8; r++) {
                int gm = m0 + wr * 32 + i * 16 + hi + r;
                C[(size_t)gm * ldc + gn] = acc[i][j][r];
            }
    }
}

// ---------------------------------------------------------------------------
// Row LayerNorm, in place. One 128-thread block per row. N <= 512.
// ---------------------------------------------------------------------------
__global__ void ln_kernel(float* __restrict__ X, const float* __restrict__ g,
                          const float* __restrict__ b, int N)
{
    __shared__ float red[128];
    const int row = blockIdx.x;
    const int t   = threadIdx.x;
    float* xr = X + (size_t)row * N;

    float lv[4], s = 0.0f;
    #pragma unroll
    for (int i = 0; i < 4; i++) {
        int c = t + i * 128;
        float v = (c < N) ? xr[c] : 0.0f;
        lv[i] = v; s += v;
    }
    red[t] = s; __syncthreads();
    for (int off = 64; off > 0; off >>= 1) {
        if (t < off) red[t] += red[t + off];
        __syncthreads();
    }
    float mean = red[0] / (float)N;
    __syncthreads();

    float s2 = 0.0f;
    #pragma unroll
    for (int i = 0; i < 4; i++) {
        int c = t + i * 128;
        float d = (c < N) ? (lv[i] - mean) : 0.0f;
        s2 += d * d;
    }
    red[t] = s2; __syncthreads();
    for (int off = 64; off > 0; off >>= 1) {
        if (t < off) red[t] += red[t + off];
        __syncthreads();
    }
    float rstd = rsqrtf(red[0] / (float)N + 1e-5f);
    #pragma unroll
    for (int i = 0; i < 4; i++) {
        int c = t + i * 128;
        if (c < N) xr[c] = (lv[i] - mean) * rstd * g[c] + b[c];
    }
}

// ---------------------------------------------------------------------------
// Build bf16 per-head q/k (RoPE, zero-pad 72->96, 1/sqrt(72) folded into q)
// and pre-transposed v [bh][d=64][s]. grid = (S, B*H), block = 128.
// ---------------------------------------------------------------------------
__global__ void pack_heads_kernel(const float* __restrict__ Q,
                                  const float* __restrict__ Qr,
                                  const float* __restrict__ KV,
                                  const float* __restrict__ Kr,
                                  unsigned short* __restrict__ qhd,
                                  unsigned short* __restrict__ khd,
                                  unsigned short* __restrict__ vht)
{
    const int s  = blockIdx.x;
    const int bh = blockIdx.y;
    const int b  = bh >> 3, h = bh & 7;
    const int j  = threadIdx.x;
    const size_t srcrow = (size_t)(b * S_LEN + s);
    const size_t dstrow = (size_t)bh * S_LEN + s;
    const float qscale = rsqrtf((float)(DHEAD + RDIM));

    if (j < HPAD) {
        float qv = 0.0f, kv = 0.0f;
        if (j < DHEAD) {
            qv = Q[srcrow * 512 + h * DHEAD + j];
            kv = KV[srcrow * 1024 + h * DHEAD + j];
        } else if (j < DHEAD + RDIM) {
            int r = j - DHEAD;
            int i = r & 3;
            float invf = __powf(10000.0f, -(float)(2 * i) / (float)RDIM);
            float ang  = (float)s * invf;
            float cs = __cosf(ang), sn = __sinf(ang);
            const float* qrp = Qr + srcrow * 64 + h * RDIM;
            const float* krp = Kr + srcrow * 64 + h * RDIM;
            float rotq = (r < 4) ? -qrp[r + 4] : qrp[r - 4];
            float rotk = (r < 4) ? -krp[r + 4] : krp[r - 4];
            qv = qrp[r] * cs + rotq * sn;
            kv = krp[r] * cs + rotk * sn;
        }
        qhd[dstrow * HPAD + j] = f2bf(qv * qscale);
        khd[dstrow * HPAD + j] = f2bf(kv);
    }
    if (j < DHEAD)
        vht[((size_t)bh * DHEAD + j) * S_LEN + s] =
            f2bf(KV[srcrow * 1024 + 512 + h * DHEAD + j]);
}

// ---------------------------------------------------------------------------
// Causal flash attention, bf16 WMMA, async double-buffered kv tiles.
// grid = (S/64, B*H), block = 128 (4 waves, 16 q-rows each), 32-key kv tiles.
// Per kv tile: 6 QK^T wmma + 4 PV wmma. Output written as bf16.
// ---------------------------------------------------------------------------
__global__ void mla_attn_kernel(const unsigned short* __restrict__ qhd,
                                const unsigned short* __restrict__ khd,
                                const unsigned short* __restrict__ vht,
                                unsigned short* __restrict__ ao)
{
    __shared__ __align__(16) unsigned short sk[2][32 * HPAD];   // [key][d]
    __shared__ __align__(16) unsigned short sv[2][64 * 32];     // [d][key]
    __shared__ __align__(16) unsigned short sp[4][16 * 32];     // per-wave P

    const int t    = threadIdx.x;
    const int lane = t & 31;
    const int wid  = t >> 5;
    const int bh   = blockIdx.y;
    const int b    = bh >> 3, h = bh & 7;
    const int qb0  = blockIdx.x * 64;
    const int aRow = lane & 15;
    const int hi   = ((lane >> 4) & 1) * 8;

    // persistent q fragments: direct 16B global loads of bf16 runs
    const int growA = qb0 + wid * 16 + aRow;
    const unsigned short* qp = qhd + ((size_t)bh * S_LEN + growA) * HPAD;
    FragAB fq[3];
    #pragma unroll
    for (int c = 0; c < 3; c++) {
        fq[c].h[0] = *(const u16x8*)(qp + c * 32 + hi);
        fq[c].h[1] = *(const u16x8*)(qp + c * 32 + 16 + hi);
    }

    auto issue_kv = [&](int kv0, int buf) {
        #pragma unroll
        for (int i = 0; i < 3; i++) {           // k: 32 rows x 12 chunks
            int id = i * 128 + t;
            int r = id / 12, ch = id % 12;
            async_b128(khd + ((size_t)bh * S_LEN + kv0 + r) * HPAD + ch * 8,
                       &sk[buf][r * HPAD + ch * 8]);
        }
        #pragma unroll
        for (int i = 0; i < 2; i++) {           // v: 64 d-rows x 4 chunks
            int id = i * 128 + t;
            int r = id >> 2, ch = id & 3;
            async_b128(vht + ((size_t)bh * DHEAD + r) * S_LEN + kv0 + ch * 8,
                       &sv[buf][r * 32 + ch * 8]);
        }
    };

    float mrun[8], lrun[8];
    v8f acc[4];
    #pragma unroll
    for (int r = 0; r < 8; r++) { mrun[r] = -3.0e38f; lrun[r] = 0.0f; }
    #pragma unroll
    for (int n = 0; n < 4; n++) acc[n] = zero8();

    const int qhiRow = qb0 + 63;
    issue_kv(0, 0);
    for (int kv0 = 0; kv0 <= qhiRow; kv0 += 32) {
        if (kv0 + 32 <= qhiRow) {
            issue_kv(kv0 + 32, ((kv0 >> 5) + 1) & 1);
            WAIT_ASYNC(5);                      // current tile's 5 loads done
        } else {
            WAIT_ASYNC(0);
        }
        __syncthreads();
        const unsigned short* skb = sk[(kv0 >> 5) & 1];
        const unsigned short* svb = sv[(kv0 >> 5) & 1];

        // scores: two 16-key sub-tiles over 3 d-chunks
        v8f s0 = zero8(), s1 = zero8();
        #pragma unroll
        for (int c = 0; c < 3; c++) {
            v16bf kb0 = load_frag(skb, aRow,      HPAD, c * 32, lane);
            v16bf kb1 = load_frag(skb, 16 + aRow, HPAD, c * 32, lane);
            s0 = wmma_bf16(fq[c].v, kb0, s0);
            s1 = wmma_bf16(fq[c].v, kb1, s1);
        }

        // causal mask + online softmax
        const int col0 = kv0 + (lane & 15);
        const int col1 = col0 + 16;
        float p0[8], p1[8];
        #pragma unroll
        for (int r = 0; r < 8; r++) {
            int gr = qb0 + wid * 16 + hi + r;
            float a0 = (col0 <= gr) ? s0[r] : -3.0e38f;
            float a1 = (col1 <= gr) ? s1[r] : -3.0e38f;
            float mx = fmaxf(a0, a1);
            mx = fmaxf(mx, __shfl_xor(mx, 1, 32));
            mx = fmaxf(mx, __shfl_xor(mx, 2, 32));
            mx = fmaxf(mx, __shfl_xor(mx, 4, 32));
            mx = fmaxf(mx, __shfl_xor(mx, 8, 32));
            float mnew  = fmaxf(mrun[r], mx);
            float alpha = __expf(mrun[r] - mnew);
            float e0 = __expf(a0 - mnew);
            float e1 = __expf(a1 - mnew);
            float rs = e0 + e1;
            rs += __shfl_xor(rs, 1, 32);
            rs += __shfl_xor(rs, 2, 32);
            rs += __shfl_xor(rs, 4, 32);
            rs += __shfl_xor(rs, 8, 32);
            lrun[r] = lrun[r] * alpha + rs;
            mrun[r] = mnew;
            p0[r] = e0; p1[r] = e1;
            #pragma unroll
            for (int n = 0; n < 4; n++) acc[n][r] *= alpha;
        }

        // P: C-layout regs -> A-layout via wave-private LDS (in-order DS pipe)
        unsigned short* spw = sp[wid];
        #pragma unroll
        for (int r = 0; r < 8; r++) {
            spw[(hi + r) * 32 + (lane & 15)]      = f2bf(p0[r]);
            spw[(hi + r) * 32 + 16 + (lane & 15)] = f2bf(p1[r]);
        }
        WAIT_DS0();

        v16bf fp = load_frag(spw, aRow, 32, 0, lane);
        #pragma unroll
        for (int n = 0; n < 4; n++) {
            v16bf fv = load_frag(svb, n * 16 + aRow, 32, 0, lane);
            acc[n] = wmma_bf16(fp, fv, acc[n]);
        }
        __syncthreads();
    }

    // epilogue: normalize, emit bf16 heads packed back to [b, s, 512]
    #pragma unroll
    for (int n = 0; n < 4; n++)
        #pragma unroll
        for (int r = 0; r < 8; r++) {
            int gr = qb0 + wid * 16 + hi + r;
            float o = acc[n][r] / lrun[r];
            ao[((size_t)b * S_LEN + gr) * 512 + h * DHEAD + n * 16 +
               (lane & 15)] = f2bf(o);
        }
}

// ---------------------------------------------------------------------------
extern "C" void kernel_launch(void* const* d_in, const int* in_sizes, int n_in,
                              void* d_out, int out_size, void* d_ws,
                              size_t ws_size, hipStream_t stream)
{
    const float* x     = (const float*)d_in[0];
    const float* W_dq  = (const float*)d_in[1];
    const float* W_uq  = (const float*)d_in[2];
    const float* W_qr  = (const float*)d_in[3];
    const float* W_dkv = (const float*)d_in[4];
    const float* W_ukv = (const float*)d_in[5];
    const float* W_kr  = (const float*)d_in[6];
    const float* W_o   = (const float*)d_in[7];
    const float* q_g   = (const float*)d_in[8];
    const float* q_b   = (const float*)d_in[9];
    const float* kv_g  = (const float*)d_in[10];
    const float* kv_b  = (const float*)d_in[11];

    float* out = (float*)d_out;                       // [4096, 512]
    float* ckv = out + (size_t)BS_ROWS * 512;         // [4096, 341]

    // fp32 workspace
    float* ws = (float*)d_ws;
    size_t off = 0;
    float* cq = ws + off; off += (size_t)BS_ROWS * 256;
    float* Q  = ws + off; off += (size_t)BS_ROWS * 512;
    float* Qr = ws + off; off += (size_t)BS_ROWS * 64;
    float* KV = ws + off; off += (size_t)BS_ROWS * 1024;
    float* Kr = ws + off; off += (size_t)BS_ROWS * 64;
    // bf16 workspace
    unsigned short* hb = (unsigned short*)(ws + off);
    size_t ho = 0;
    unsigned short* x_bf   = hb + ho; ho += (size_t)BS_ROWS * 512;
    unsigned short* cq_bf  = hb + ho; ho += (size_t)BS_ROWS * 256;
    unsigned short* ckv_bf = hb + ho; ho += (size_t)BS_ROWS * 352;
    unsigned short* ao_bf  = hb + ho; ho += (size_t)BS_ROWS * 512;
    unsigned short* qhd    = hb + ho; ho += (size_t)16 * S_LEN * HPAD;
    unsigned short* khd    = hb + ho; ho += (size_t)16 * S_LEN * HPAD;
    unsigned short* vht    = hb + ho; ho += (size_t)16 * DHEAD * S_LEN;
    unsigned short* wdq_t  = hb + ho; ho += (size_t)256  * 512;
    unsigned short* wdkv_t = hb + ho; ho += (size_t)384  * 512;
    unsigned short* wkr_t  = hb + ho; ho += (size_t)128  * 512;
    unsigned short* wuq_t  = hb + ho; ho += (size_t)512  * 256;
    unsigned short* wqr_t  = hb + ho; ho += (size_t)128  * 256;
    unsigned short* wukv_t = hb + ho; ho += (size_t)1024 * 352;
    unsigned short* wo_t   = hb + ho; ho += (size_t)512  * 512;   // ~62 MB

    auto convW = [&](const float* W, unsigned short* Wt, int K, int N,
                     int Kpad, int Npad, int srcKN) {
        int total = Npad * Kpad;
        convW_kernel<<<(total + 255) / 256, 256, 0, stream>>>(W, Wt, K, N,
                                                              Kpad, total,
                                                              srcKN);
    };
    auto convA = [&](const float* X, unsigned short* Xb, int K, int Kpad) {
        int total = BS_ROWS * Kpad;
        convA_kernel<<<(total + 255) / 256, 256, 0, stream>>>(X, Xb, K, Kpad,
                                                              total);
    };
    auto gemm = [&](const unsigned short* A, const unsigned short* Bt,
                    float* C, int K, int Npad, int Nout) {
        dim3 grid(Npad / 128, BS_ROWS / 64);
        gemm_async_kernel<<<grid, 256, 0, stream>>>(A, Bt, C, K, Nout, Nout);
    };

    // one-time bf16 conversions (weights transposed+padded to [Npad][Kpad])
    convW(W_dq,  wdq_t,  512, 256,  512, 256,  1);
    convW(W_dkv, wdkv_t, 512, 341,  512, 384,  1);
    convW(W_kr,  wkr_t,  512, 64,   512, 128,  1);
    convW(W_uq,  wuq_t,  256, 512,  256, 512,  1);
    convW(W_qr,  wqr_t,  256, 64,   256, 128,  1);
    convW(W_ukv, wukv_t, 341, 1024, 352, 1024, 1);
    convW(W_o,   wo_t,   512, 512,  512, 512,  0);   // W_o^T: rows already [n][k]
    convA(x, x_bf, 512, 512);

    // down-projections
    gemm(x_bf, wdq_t,  cq,  512, 256, 256);
    gemm(x_bf, wdkv_t, ckv, 512, 384, 341);
    gemm(x_bf, wkr_t,  Kr,  512, 128, 64);
    // layernorms (ckv lives directly in d_out)
    ln_kernel<<<BS_ROWS, 128, 0, stream>>>(cq,  q_g,  q_b, 256);
    ln_kernel<<<BS_ROWS, 128, 0, stream>>>(ckv, kv_g, kv_b, 341);
    convA(cq,  cq_bf,  256, 256);
    convA(ckv, ckv_bf, 341, 352);
    // up-projections
    gemm(cq_bf,  wuq_t,  Q,  256, 512,  512);
    gemm(cq_bf,  wqr_t,  Qr, 256, 128,  64);
    gemm(ckv_bf, wukv_t, KV, 352, 1024, 1024);
    // head packing + RoPE (emits bf16 q/k and transposed v)
    pack_heads_kernel<<<dim3(S_LEN, 16), 128, 0, stream>>>(Q, Qr, KV, Kr,
                                                           qhd, khd, vht);
    // causal flash attention (bf16 in/out)
    mla_attn_kernel<<<dim3(S_LEN / 64, 16), 128, 0, stream>>>(qhd, khd, vht,
                                                              ao_bf);
    // output projection: out = ao @ W_o^T
    gemm(ao_bf, wo_t, out, 512, 512, 512);
}
